// Stress_LMSC_90787018703565
// MI455X (gfx1250) — matrix operand order repaired and checked by reading, not compiled, not run on hardware
//
#include <hip/hip_runtime.h>

// ---------------------------------------------------------------------------
// LMSC recurrent gated network on MI455X (gfx1250).
// One wave32 owns a 16-row batch tile and iterates the full T=1024 scan with
// weights resident in LDS as pre-swizzled f16 WMMA B-fragments.
// All matmuls use v_wmma_f32_16x16x32_f16.
// ---------------------------------------------------------------------------

typedef __attribute__((ext_vector_type(16))) _Float16 v16h;
typedef __attribute__((ext_vector_type(8)))  _Float16 v8h;
typedef __attribute__((ext_vector_type(8)))  float    v8f;

// Problem constants
constexpr int Bn = 1024, Tn = 1024;

// Workspace / LDS fragment layout (bytes). Each 32x16 (KxN) f16 B-fragment is
// 1024B: lane l's first 16B at frag+16*l, second 16B at frag+512+16*l.
constexpr int WS_W1_0 = 0;            // qb1_w0: 3 ktiles x 8 ntiles = 24 KB
constexpr int WS_W2_0 = 24576;        // qb2_w0: 24 KB
constexpr int WS_W1_1 = 49152;        // qb1_w1: 4x8 = 32 KB
constexpr int WS_W2_1 = 81920;
constexpr int WS_W1_2 = 114688;
constexpr int WS_W2_2 = 147456;
constexpr int WS_WA   = 180224;       // wa: 4 ktiles x 4 ntiles = 16 KB
constexpr int WS_WB   = 196608;
constexpr int WS_WOUT = 212992;       // wout: 2 ktiles x 1 ntile = 2 KB
constexpr int WS_BIAS = 215040;       // f32: 6*128 layer biases + ba(64) + bb(64)
constexpr int WS_FRAG_BYTES = 215040;

// LDS layout: [0,WS_FRAG_BYTES) = fragments; then 4 per-wave c-buffers
// (16 rows x 272B, 128 f16 cols + pad for bank spread); then 4x16 f32 norms.
constexpr int CROW       = 272;
constexpr int CBUF_OFF   = WS_FRAG_BYTES;
constexpr int NORM_OFF   = CBUF_OFF + 4 * 16 * CROW;   // 232448
constexpr int SMEM_BYTES = NORM_OFF + 4 * 64;          // 232704

__device__ __forceinline__ float fast_tanh(float x) {
  // tanh(x) = 1 - 2/(1+exp(2x)); saturates cleanly at +-1 (v_exp_f32 trans op)
  return 1.0f - 2.0f / (1.0f + __expf(2.0f * x));
}

__device__ __forceinline__ v16h ld_b_frag(const char* frag, int lane) {
  v8h lo = *(const v8h*)(frag + lane * 16);
  v8h hi = *(const v8h*)(frag + 512 + lane * 16);
  v16h r;
#pragma unroll
  for (int i = 0; i < 8; i++) { r[i] = lo[i]; r[i + 8] = hi[i]; }
  return r;
}

// A-fragment (16x32 f16) from the row-major c-buffer. crow points at this
// lane's row (m = lane&15); byteoff = kt*64 + (lane>>4)*16. Two aligned b128s.
__device__ __forceinline__ v16h ld_a_frag(const char* crow, int byteoff) {
  v8h lo = *(const v8h*)(crow + byteoff);
  v8h hi = *(const v8h*)(crow + byteoff + 32);
  v16h r;
#pragma unroll
  for (int i = 0; i < 8; i++) { r[i] = lo[i]; r[i + 8] = hi[i]; }
  return r;
}

// One gated layer: acc1 = A@W1 + b1, acc2 = A@W2 + b2, write product to c.
// MODE 0: tanh(x1)*tanh(x2) (layers 0,1). MODE 1: tanh(x1*x2) (produces g).
template <int KT, int MODE>
__device__ __forceinline__ void run_layer(const char* smem, char* cbase,
                                          const v16h* a, int base1, int base2,
                                          const float* b1, const float* b2,
                                          int lane) {
  const int m8 = lane >> 4, n16 = lane & 15;
#pragma unroll
  for (int nt = 0; nt < 8; nt++) {
    v8f acc1, acc2;
#pragma unroll
    for (int v = 0; v < 8; v++) { acc1[v] = b1[nt]; acc2[v] = b2[nt]; }
#pragma unroll
    for (int kt = 0; kt < KT; kt++)
      acc1 = __builtin_amdgcn_wmma_f32_16x16x32_f16(
          false, a[kt], false,
          ld_b_frag(smem + base1 + (kt * 8 + nt) * 1024, lane),
          (short)0, acc1, false, false);
#pragma unroll
    for (int kt = 0; kt < KT; kt++)
      acc2 = __builtin_amdgcn_wmma_f32_16x16x32_f16(
          false, a[kt], false,
          ld_b_frag(smem + base2 + (kt * 8 + nt) * 1024, lane),
          (short)0, acc2, false, false);
#pragma unroll
    for (int v = 0; v < 8; v++) {
      float p = (MODE == 0) ? fast_tanh(acc1[v]) * fast_tanh(acc2[v])
                            : fast_tanh(acc1[v] * acc2[v]);
      *(_Float16*)(cbase + (v + 8 * m8) * CROW + (nt * 16 + n16) * 2) =
          (_Float16)p;
    }
  }
}

__global__ __launch_bounds__(128, 1) void lmsc_main(
    const float* __restrict__ x, const float* __restrict__ initF,
    const char* __restrict__ ws, float* __restrict__ out) {
  extern __shared__ char smem[];
  const int lane = threadIdx.x & 31, wave = threadIdx.x >> 5;
  const int m8 = lane >> 4, n16 = lane & 15;
  const int r0 = (blockIdx.x * 4 + wave) * 16;   // batch row base of this wave

  // Stage all weight fragments into LDS once (reused 1024 steps).
  {
    const uint4* s = (const uint4*)ws;
    uint4* d = (uint4*)smem;
    for (int i = threadIdx.x; i < WS_FRAG_BYTES / 16; i += 128) d[i] = s[i];
  }
  __syncthreads();

  char* cbase = smem + CBUF_OFF + wave * (16 * CROW);
  float* nb   = (float*)(smem + NORM_OFF + wave * 64);
  const float* bias = (const float*)(ws + WS_BIAS);

  // Per-lane bias scalars (depend only on output column = nt*16+n16).
  float bi1[3][8], bi2[3][8], bA[4], bB[4];
#pragma unroll
  for (int i = 0; i < 3; i++)
#pragma unroll
    for (int nt = 0; nt < 8; nt++) {
      bi1[i][nt] = bias[i * 256 + nt * 16 + n16];
      bi2[i][nt] = bias[i * 256 + 128 + nt * 16 + n16];
    }
#pragma unroll
  for (int nt = 0; nt < 4; nt++) {
    bA[nt] = bias[768 + nt * 16 + n16];
    bB[nt] = bias[832 + nt * 16 + n16];
  }

  // Hidden state in WMMA C-layout: h[nt][v] = h[row v+8*m8][col nt*16+n16].
  float h[4][8];
#pragma unroll
  for (int nt = 0; nt < 4; nt++)
#pragma unroll
    for (int v = 0; v < 8; v++) {
      h[nt][v] = initF[(r0 + v + 8 * m8) * 66 + 2 + nt * 16 + n16];
      *(_Float16*)(cbase + (v + 8 * m8) * CROW + (8 + nt * 16 + n16) * 2) =
          (_Float16)h[nt][v];
    }

  const int AB = Bn * Tn * 6;                     // alphas base in d_out
  const char* crow = cbase + (lane & 15) * CROW;  // this lane's A row

  for (int t = 0; t < Tn; t++) {
    // --- stage a: norm + xn + zero padding (cols 0..7, 72..95) ---
    if (lane < 16) {
      const float* xp = x + ((r0 + lane) * Tn + t) * 6;
      float x0 = xp[0], x1 = xp[1], x2 = xp[2], x3 = xp[3], x4 = xp[4],
            x5 = xp[5];
      float s = x0 * x0 + x1 * x1 + x2 * x2 + x3 * x3 + x4 * x4 + x5 * x5;
      float nrm = sqrtf(s);
      nb[lane] = nrm;
      float inv = 1.0f / (nrm + 1e-8f);
      char* row = cbase + lane * CROW;
      _Float16* rh = (_Float16*)row;
      rh[0] = (_Float16)(x0 * inv); rh[1] = (_Float16)(x1 * inv);
      rh[2] = (_Float16)(x2 * inv); rh[3] = (_Float16)(x3 * inv);
      rh[4] = (_Float16)(x4 * inv); rh[5] = (_Float16)(x5 * inv);
      rh[6] = (_Float16)0.0f;       rh[7] = (_Float16)0.0f;
      float4 z = make_float4(0.f, 0.f, 0.f, 0.f);
      *(float4*)(row + 144) = z;  // cols 72..79
      *(float4*)(row + 160) = z;  // cols 80..87
      *(float4*)(row + 176) = z;  // cols 88..95
    }
    // (same-wave LDS ops are in order on CDNA5; no barrier needed)

    v16h a[4];
    // layer 0: K = 96 padded
#pragma unroll
    for (int kt = 0; kt < 3; kt++) a[kt] = ld_a_frag(crow, kt * 64 + m8 * 16);
    run_layer<3, 0>(smem, cbase, a, WS_W1_0, WS_W2_0, bi1[0], bi2[0], lane);
    // layer 1
#pragma unroll
    for (int kt = 0; kt < 4; kt++) a[kt] = ld_a_frag(crow, kt * 64 + m8 * 16);
    run_layer<4, 0>(smem, cbase, a, WS_W1_1, WS_W2_1, bi1[1], bi2[1], lane);
    // layer 2 -> writes g = tanh(x1*x2)
#pragma unroll
    for (int kt = 0; kt < 4; kt++) a[kt] = ld_a_frag(crow, kt * 64 + m8 * 16);
    run_layer<4, 1>(smem, cbase, a, WS_W1_2, WS_W2_2, bi1[2], bi2[2], lane);

    // --- heads: alpha = exp(g@wa+ba), beta = tanh(g@wb+bb), h update ---
#pragma unroll
    for (int kt = 0; kt < 4; kt++) a[kt] = ld_a_frag(crow, kt * 64 + m8 * 16);
    float nrm8[8];
    *(float4*)&nrm8[0] = *(const float4*)(nb + 8 * m8);
    *(float4*)&nrm8[4] = *(const float4*)(nb + 8 * m8 + 4);
#pragma unroll
    for (int nt = 0; nt < 4; nt++) {
      v8f accA, accB;
#pragma unroll
      for (int v = 0; v < 8; v++) { accA[v] = bA[nt]; accB[v] = bB[nt]; }
#pragma unroll
      for (int kt = 0; kt < 4; kt++)
        accA = __builtin_amdgcn_wmma_f32_16x16x32_f16(
            false, a[kt], false,
            ld_b_frag(smem + WS_WA + (kt * 4 + nt) * 1024, lane),
            (short)0, accA, false, false);
#pragma unroll
      for (int kt = 0; kt < 4; kt++)
        accB = __builtin_amdgcn_wmma_f32_16x16x32_f16(
            false, a[kt], false,
            ld_b_frag(smem + WS_WB + (kt * 4 + nt) * 1024, lane),
            (short)0, accB, false, false);
#pragma unroll
      for (int v = 0; v < 8; v++) {
        float al = __expf(accA[v]);
        float be = fast_tanh(accB[v]);
        int b = r0 + v + 8 * m8;
        out[AB + (b * Tn + t) * 64 + nt * 16 + n16] = al;
        float hn = __expf(-al * nrm8[v]) * (h[nt][v] - be) + be;
        h[nt][v] = hn;
        // write h_new to c cols 8..71 (also next step's h input)
        *(_Float16*)(cbase + (v + 8 * m8) * CROW + (8 + nt * 16 + n16) * 2) =
            (_Float16)hn;
      }
    }

    // --- out = h_new @ wout (K=64 over c cols 8..71, N padded to 16) ---
    a[0] = ld_a_frag(crow, 16 + m8 * 16);
    a[1] = ld_a_frag(crow, 80 + m8 * 16);
    v8f ao;
#pragma unroll
    for (int v = 0; v < 8; v++) ao[v] = 0.0f;
    ao = __builtin_amdgcn_wmma_f32_16x16x32_f16(
        false, a[0], false, ld_b_frag(smem + WS_WOUT, lane), (short)0, ao,
        false, false);
    ao = __builtin_amdgcn_wmma_f32_16x16x32_f16(
        false, a[1], false, ld_b_frag(smem + WS_WOUT + 1024, lane), (short)0,
        ao, false, false);
    if (n16 < 6) {
#pragma unroll
      for (int v = 0; v < 8; v++) {
        int b = r0 + v + 8 * m8;
        out[(b * Tn + t) * 6 + n16] = ao[v];
      }
    }
  }
}

// ---------------------------------------------------------------------------
// Repack f32 weights -> swizzled f16 WMMA B-fragments + f32 bias table in ws.
// B-fragment (32x16 KxN) lane l = (khalf=l>>4, n=l&15); element j (0..15)
// holds W[kt*32 + (j<8 ? khalf*8 + j : 16 + khalf*8 + j-8)][nt*16 + n].
// ---------------------------------------------------------------------------
__global__ void lmsc_repack(const float* w10, const float* w20,
                            const float* w11, const float* w21,
                            const float* w12, const float* w22,
                            const float* wa, const float* wb,
                            const float* wo, const float* b10,
                            const float* b20, const float* b11,
                            const float* b21, const float* b12,
                            const float* b22, const float* ba,
                            const float* bb, char* ws) {
  int tid = blockIdx.x * blockDim.x + threadIdx.x;
  if (tid < 210 * 32) {
    int f = tid >> 5, lane = tid & 31;
    const int starts[10] = {0, 24, 48, 80, 112, 144, 176, 192, 208, 210};
    const int nts[9]   = {8, 8, 8, 8, 8, 8, 4, 4, 1};
    const int offs[9]  = {WS_W1_0, WS_W2_0, WS_W1_1, WS_W2_1, WS_W1_2,
                          WS_W2_2, WS_WA, WS_WB, WS_WOUT};
    const float* Ws[9] = {w10, w20, w11, w21, w12, w22, wa, wb, wo};
    const int modes[9] = {0, 0, 1, 1, 1, 1, 2, 2, 3};
    int mi = 0;
    while (f >= starts[mi + 1]) mi++;
    int fl = f - starts[mi];
    int NT = nts[mi], kt = fl / NT, nt = fl % NT;
    int khalf = lane >> 4, n16 = lane & 15, n_g = nt * 16 + n16;
    const float* W = Ws[mi];
    int mode = modes[mi];
    char* frag = ws + offs[mi] + fl * 1024;
    for (int j = 0; j < 16; j++) {
      int kpad = kt * 32 + ((j < 8) ? (khalf * 8 + j) : (16 + khalf * 8 + (j - 8)));
      float v = 0.0f;
      if (mode == 0) {
        // layer0 padded rows: 0..5 = xn rows, 8..71 = h rows, rest zero
        int r = (kpad < 6) ? kpad : ((kpad >= 8 && kpad < 72) ? (kpad - 2) : -1);
        if (r >= 0) v = W[r * 128 + n_g];
      } else if (mode == 1) {
        v = W[kpad * 128 + n_g];                   // 128x128
      } else if (mode == 2) {
        v = W[kpad * 64 + n_g];                    // 128x64
      } else {
        if (n_g < 6) v = W[kpad * 6 + n_g];        // 64x6 (N padded to 16)
      }
      int boff = (j < 8) ? (lane * 16 + j * 2) : (512 + lane * 16 + (j - 8) * 2);
      *(_Float16*)(frag + boff) = (_Float16)v;
    }
  } else if (tid < 210 * 32 + 896) {
    int i = tid - 210 * 32;
    const float* Bs[8] = {b10, b20, b11, b21, b12, b22, ba, bb};
    float v;
    if (i < 768)      v = Bs[i / 128][i % 128];
    else if (i < 832) v = ba[i - 768];
    else              v = bb[i - 832];
    ((float*)(ws + WS_BIAS))[i] = v;
  }
}

extern "C" void kernel_launch(void* const* d_in, const int* in_sizes, int n_in,
                              void* d_out, int out_size, void* d_ws,
                              size_t ws_size, hipStream_t stream) {
  (void)in_sizes; (void)n_in; (void)out_size; (void)ws_size;
  const float* x     = (const float*)d_in[0];
  const float* initF = (const float*)d_in[1];
  const float* w10   = (const float*)d_in[2];
  const float* b10   = (const float*)d_in[3];
  const float* w20   = (const float*)d_in[4];
  const float* b20   = (const float*)d_in[5];
  const float* w11   = (const float*)d_in[6];
  const float* b11   = (const float*)d_in[7];
  const float* w21   = (const float*)d_in[8];
  const float* b21   = (const float*)d_in[9];
  const float* w12   = (const float*)d_in[10];
  const float* b12   = (const float*)d_in[11];
  const float* w22   = (const float*)d_in[12];
  const float* b22   = (const float*)d_in[13];
  const float* wa    = (const float*)d_in[14];
  const float* ba    = (const float*)d_in[15];
  const float* wb    = (const float*)d_in[16];
  const float* bb    = (const float*)d_in[17];
  const float* wo    = (const float*)d_in[18];
  char* ws = (char*)d_ws;
  float* out = (float*)d_out;

  (void)hipFuncSetAttribute(reinterpret_cast<const void*>(lmsc_main),
                            hipFuncAttributeMaxDynamicSharedMemorySize,
                            SMEM_BYTES);

  lmsc_repack<<<30, 256, 0, stream>>>(w10, w20, w11, w21, w12, w22, wa, wb, wo,
                                      b10, b20, b11, b21, b12, b22, ba, bb, ws);
  lmsc_main<<<16, 128, SMEM_BYTES, stream>>>(x, initF, ws, out);
}
